// MultiHeadAttention_73452530696515
// MI455X (gfx1250) — compile-verified, matrix operand order checked by reading, not run on hardware
//
#include <hip/hip_runtime.h>

// ---------------------------------------------------------------------------
// MHA forward (ALiBi) for MI455X / gfx1250, wave32 + v_wmma_f32_16x16x32_bf16
// B=4, T=2048, C=1024, H=16, D=64
// Fused attention: per-block 32-query-row score slab lives entirely in the
// 320 KB WGP LDS, so raw scores never round-trip HBM (saves ~3.2 GB).
// Pure-copy tile loads use GLOBAL_LOAD_ASYNC_TO_LDS (ASYNCcnt DMA path).
// ---------------------------------------------------------------------------

#define BB 4
#define TT 2048
#define CC 1024
#define HH 16
#define DD 64
#define MM (BB * TT) // 8192

#define SROW 2052    // padded scores row stride in floats (2052*4B: +4 dw/row bank rotation)

typedef __attribute__((ext_vector_type(16))) __bf16 v16bf;
typedef __attribute__((ext_vector_type(8))) float v8f;
typedef __attribute__((ext_vector_type(2))) float vf2;
typedef __attribute__((ext_vector_type(4))) float vf4;
typedef __attribute__((ext_vector_type(4))) unsigned vu4;

struct FragBF {
    union {
        unsigned u[8];
        vu4 q[2];
        v16bf v;
    };
};

static __device__ inline unsigned short f2bf(float f) {
    // round-to-nearest-even fp32 -> bf16
    unsigned u = __float_as_uint(f);
    u += 0x7FFFu + ((u >> 16) & 1u);
    return (unsigned short)(u >> 16);
}

static __device__ inline unsigned pack_bf2(float lo, float hi) {
    return (unsigned)f2bf(lo) | ((unsigned)f2bf(hi) << 16);
}

static __device__ inline v8f zero8() {
    v8f z = {0.f, 0.f, 0.f, 0.f, 0.f, 0.f, 0.f, 0.f};
    return z;
}

// Async DMA: 16B global -> LDS, tracked by ASYNCcnt (CDNA5 §10.7/§15.18).
// VDST VGPR carries the LDS byte address (low 32 bits of the generic pointer
// are the LDS offset per ISA §10.2 aperture rules).
static __device__ inline void async_ld_b128(const void* gaddr, void* laddr) {
    asm volatile("global_load_async_to_lds_b128 %0, %1, off"
                 :
                 : "v"((unsigned)(size_t)laddr), "v"(gaddr)
                 : "memory");
}
static __device__ inline void wait_async0() {
    asm volatile("s_wait_asynccnt 0x0" ::: "memory");
}

// Load a 16x32 bf16 fragment from LDS stored as uint32 bf16-pairs.
// Per CDNA5 ISA 7.12.2: pair index for VGPR v:
//   p = pbase + (v>=4 ? 8 : 0) + (v&3) + 4*(lane>>4)
// -> u[0..3] / u[4..7] contiguous: two ds_load_b128. LDP multiple of 4.
template <int LDP>
static __device__ inline void load_frag(FragBF& f, const unsigned* lds, int base,
                                        int lane, int pbase) {
    const int lo = lane & 15;
    const int hi = lane >> 4;
    const unsigned* ptr = lds + (base + lo) * LDP + pbase + (hi << 2);
    f.q[0] = *reinterpret_cast<const vu4*>(ptr);
    f.q[1] = *reinterpret_cast<const vu4*>(ptr + 8);
}

static __device__ inline v8f wmma_bf16(const FragBF& a, const FragBF& b, v8f c) {
    return __builtin_amdgcn_wmma_f32_16x16x32_bf16(
        /*neg_a=*/false, a.v, /*neg_b=*/false, b.v,
        /*c_mod=*/(short)0, c, /*reuse_a=*/false, /*reuse_b=*/false);
}

// ---------------------------------------------------------------------------
// Kernel 1: P = bf16(x @ W + bias), output in [B,H,T,D] layout.
// Grid: (N/128=8, M/128=64), 256 threads (8 waves, 2x4 wave grid, 64x32/wave)
// (A/B need fp32->bf16 conversion, so these loaders stay on the VALU path.)
// ---------------------------------------------------------------------------
__global__ __launch_bounds__(256) void qkv_proj(const float* __restrict__ x,
                                                const float* __restrict__ W,
                                                const float* __restrict__ bias,
                                                unsigned short* __restrict__ out) {
    __shared__ unsigned lds_a[128 * 20];
    __shared__ unsigned lds_b[128 * 20];
    const int tid = threadIdx.x;
    const int lane = tid & 31, wave = tid >> 5;
    const int wr = wave >> 2, wc = wave & 3;
    const int m0 = blockIdx.y * 128;
    const int n0 = blockIdx.x * 128;

    v8f acc[4][2];
#pragma unroll
    for (int mt = 0; mt < 4; ++mt)
#pragma unroll
        for (int nt = 0; nt < 2; ++nt) acc[mt][nt] = zero8();

    for (int k0 = 0; k0 < CC; k0 += 32) {
        if (k0 + 32 < CC) {  // overlap next tile's HBM fetch with this compute
            __builtin_prefetch(x + (size_t)(m0 + (tid >> 4)) * CC + k0 + 32, 0, 3);
            __builtin_prefetch(W + (size_t)(k0 + 32 + 2 * (tid >> 7)) * CC + n0 + (tid & 127), 0, 3);
        }
        // A tile: 128 rows x 16 k-pairs, fp32 -> bf16 pack
#pragma unroll
        for (int i = 0; i < 8; ++i) {
            const int idx = tid + i * 256;
            const int row = idx >> 4, p = idx & 15;
            const vf2 f2 = *reinterpret_cast<const vf2*>(
                x + (size_t)(m0 + row) * CC + k0 + 2 * p);
            lds_a[row * 20 + p] = pack_bf2(f2.x, f2.y);
        }
        // B tile: lds_b[n][p] = pack(W[k0+2p][n0+n], W[k0+2p+1][n0+n])
#pragma unroll
        for (int i = 0; i < 8; ++i) {
            const int idx = tid + i * 256;
            const int p = idx >> 7, n = idx & 127;
            const float w0 = W[(size_t)(k0 + 2 * p) * CC + n0 + n];
            const float w1 = W[(size_t)(k0 + 2 * p + 1) * CC + n0 + n];
            lds_b[n * 20 + p] = pack_bf2(w0, w1);
        }
        __syncthreads();

        FragBF bfr[2];
        load_frag<20>(bfr[0], lds_b, wc * 32 + 0, lane, 0);
        load_frag<20>(bfr[1], lds_b, wc * 32 + 16, lane, 0);
#pragma unroll
        for (int mt = 0; mt < 4; ++mt) {
            FragBF afr;
            load_frag<20>(afr, lds_a, wr * 64 + mt * 16, lane, 0);
            acc[mt][0] = wmma_bf16(afr, bfr[0], acc[mt][0]);
            acc[mt][1] = wmma_bf16(afr, bfr[1], acc[mt][1]);
        }
        __syncthreads();
    }

    const int lo = lane & 15, hi = lane >> 4;
#pragma unroll
    for (int nt = 0; nt < 2; ++nt) {
        const int n = n0 + wc * 32 + nt * 16 + lo;
        const float bv = bias[n];
        const int h = n >> 6, d = n & 63;
#pragma unroll
        for (int mt = 0; mt < 4; ++mt) {
#pragma unroll
            for (int r = 0; r < 8; ++r) {
                const int m = m0 + wr * 64 + mt * 16 + r + hi * 8;
                const int b = m >> 11, t = m & (TT - 1);
                const float val = acc[mt][nt][r] + bv;
                out[(((size_t)(b * HH + h) * TT + t) << 6) + d] = f2bf(val);
            }
        }
    }
}

// ---------------------------------------------------------------------------
// Kernel 2 (fused attention): per block, 32 query rows of one (b,h).
//   Phase 1: scores = scale*Q K^T + alibi  -> fp32 slab in LDS (32 x 2048)
//   Phase 2: row softmax in LDS; single NT b128 store of normalized attn
//   Phase 3: O = e @ V from LDS, scaled by 1/rowsum; bf16 head-out
// Q/K tiles arrive via GLOBAL_LOAD_ASYNC_TO_LDS_B128 (ASYNCcnt DMA).
// Dynamic LDS: scores 32*SROW + Qtile 32*36 + K/V tile 128*36 dwords (~279 KB)
// Grid: (T/32 = 64, B*H = 64), 256 threads.
// ---------------------------------------------------------------------------
__global__ __launch_bounds__(256) void attn_fused(const unsigned short* __restrict__ Q,
                                                  const unsigned short* __restrict__ Km,
                                                  const unsigned short* __restrict__ V,
                                                  float* __restrict__ attn,
                                                  unsigned short* __restrict__ ohead) {
    extern __shared__ unsigned smem[];
    float* sc = reinterpret_cast<float*>(smem);          // [32][SROW] fp32
    unsigned* qt = smem + 32 * SROW;                     // [32][36] bf16 pairs
    unsigned* kvt = qt + 32 * 36;                        // [128][36] K pairs / [64][68] V pairs

    __shared__ float red[256];
    __shared__ float rowm[32];
    __shared__ float rowinv[32];

    const int tid = threadIdx.x;
    const int lane = tid & 31, wave = tid >> 5;
    const int lo = lane & 15, hi = lane >> 4;
    const int zi = blockIdx.y;                           // b*H + h
    const int h = zi & (HH - 1), b = zi >> 4;
    const int q0 = blockIdx.x * 32;

    const unsigned* Qp = reinterpret_cast<const unsigned*>(Q) + (size_t)zi * TT * 32;
    const unsigned* Kp = reinterpret_cast<const unsigned*>(Km) + (size_t)zi * TT * 32;
    const unsigned short* Vh = V + (size_t)zi * TT * DD;

    // ---- Q tile: 32 rows x 32 pairs, async DMA (one b128 per thread) ----
    {
        const int row = tid >> 3, c4 = (tid & 7) * 4;
        async_ld_b128(Qp + (size_t)(q0 + row) * 32 + c4, qt + row * 36 + c4);
    }
    wait_async0();
    __syncthreads();

    // persistent Q fragments: 2 row-tiles x 2 k-halves (held through phase 1)
    FragBF qf[2][2];
#pragma unroll
    for (int mt = 0; mt < 2; ++mt)
#pragma unroll
        for (int s = 0; s < 2; ++s) load_frag<36>(qf[mt][s], qt, mt * 16, lane, s * 16);

    const float slope = exp2f(-(float)(h + 1));          // 0.5^(h+1)
    const float scale = 0.125f;                          // 1/sqrt(64)

    // ================= Phase 1: scores -> LDS =================
    for (int c0 = 0; c0 < TT; c0 += 128) {
        // K tile: 128 keys x 32 pairs, async DMA (4 b128 per thread)
#pragma unroll
        for (int i = 0; i < 4; ++i) {
            const int t = tid + i * 256;                 // 0..1023
            const int row = t >> 3, c4 = (t & 7) * 4;
            async_ld_b128(Kp + (size_t)(c0 + row) * 32 + c4, kvt + row * 36 + c4);
        }
        wait_async0();
        __syncthreads();

        // wave handles keys [c0 + wave*16, +16), all 32 q rows
        v8f a0 = zero8(), a1 = zero8();
        FragBF kf;
        load_frag<36>(kf, kvt, wave * 16, lane, 0);
        a0 = wmma_bf16(qf[0][0], kf, a0);
        a1 = wmma_bf16(qf[1][0], kf, a1);
        load_frag<36>(kf, kvt, wave * 16, lane, 16);
        a0 = wmma_bf16(qf[0][1], kf, a0);
        a1 = wmma_bf16(qf[1][1], kf, a1);

        const int col = c0 + wave * 16 + lo;
#pragma unroll
        for (int r = 0; r < 8; ++r) {
            const int r0 = r + hi * 8;
            const int r1 = 16 + r + hi * 8;
            sc[r0 * SROW + col] = a0[r] * scale + slope * (float)(col - (q0 + r0));
            sc[r1 * SROW + col] = a1[r] * scale + slope * (float)(col - (q0 + r1));
        }
        __syncthreads();  // kvt reused next chunk
    }

    // ================= Phase 2: softmax + NT attn store =================
    {
        const int row = tid >> 3;            // 8 threads per row
        const int cb = (tid & 7) * 4;        // interleaved vf4 chunks (bank-clean)
        float* rp = sc + row * SROW;

        float mx = -3.4028235e38f;
#pragma unroll 8
        for (int i = 0; i < 64; ++i) {
            const vf4 v = *reinterpret_cast<const vf4*>(rp + cb + i * 32);
            mx = fmaxf(mx, fmaxf(fmaxf(v.x, v.y), fmaxf(v.z, v.w)));
        }
        red[tid] = mx;
        __syncthreads();
        if ((tid & 7) == 0) {
            float m = red[tid];
#pragma unroll
            for (int j = 1; j < 8; ++j) m = fmaxf(m, red[tid + j]);
            rowm[row] = m;
        }
        __syncthreads();

        const float m = rowm[row];
        float sum = 0.f;
#pragma unroll 8
        for (int i = 0; i < 64; ++i) {
            vf4 v = *reinterpret_cast<const vf4*>(rp + cb + i * 32);
            v.x = __expf(v.x - m); v.y = __expf(v.y - m);
            v.z = __expf(v.z - m); v.w = __expf(v.w - m);
            sum += (v.x + v.y) + (v.z + v.w);
            *reinterpret_cast<vf4*>(rp + cb + i * 32) = v;   // keep e in LDS
        }
        red[tid] = sum;
        __syncthreads();
        if ((tid & 7) == 0) {
            float s = red[tid];
#pragma unroll
            for (int j = 1; j < 8; ++j) s += red[tid + j];
            rowinv[row] = 1.0f / s;
        }
        __syncthreads();

        // the ONLY HBM traffic for scores: normalized attn, streamed NT
        const float inv = rowinv[row];
        float* gout = attn + ((size_t)zi * TT + q0 + row) * TT;
#pragma unroll 8
        for (int i = 0; i < 64; ++i) {
            vf4 v = *reinterpret_cast<const vf4*>(rp + cb + i * 32);
            v *= inv;
            __builtin_nontemporal_store(v, reinterpret_cast<vf4*>(gout + cb + i * 32));
        }
    }
    __syncthreads();

    // ================= Phase 3: O = e @ V (scaled by rowinv) =================
    // wave -> (row tile mt3, 16-wide d strip ns3); 1 accumulator per wave
    const int mt3 = wave >> 2;
    const int ns3 = (wave & 3) * 16;
    v8f oacc = zero8();

    for (int c0 = 0; c0 < TT; c0 += 128) {
        // V tile transposed to [d(64)][kpair(64)] bf16 pairs, stride 68
        // (transpose means this one stays on the VALU pack path)
#pragma unroll
        for (int i = 0; i < 16; ++i) {
            const int t = tid + i * 256;
            const int d = t & 63, pk = t >> 6;
            const unsigned short w0 = Vh[(size_t)(c0 + 2 * pk) * DD + d];
            const unsigned short w1 = Vh[(size_t)(c0 + 2 * pk + 1) * DD + d];
            kvt[d * 68 + pk] = (unsigned)w0 | ((unsigned)w1 << 16);
        }
        __syncthreads();

#pragma unroll
        for (int kk = 0; kk < 4; ++kk) {
            // A fragment: e rows, k = c0 + kk*32 .. +31, fp32 LDS -> bf16 pack
            FragBF af;
            {
                const float* rp = sc + (mt3 * 16 + lo) * SROW + c0 + kk * 32 + hi * 8;
                const vf4 f0 = *reinterpret_cast<const vf4*>(rp);
                const vf4 f1 = *reinterpret_cast<const vf4*>(rp + 4);
                const vf4 f2 = *reinterpret_cast<const vf4*>(rp + 16);
                const vf4 f3 = *reinterpret_cast<const vf4*>(rp + 20);
                af.u[0] = pack_bf2(f0.x, f0.y);
                af.u[1] = pack_bf2(f0.z, f0.w);
                af.u[2] = pack_bf2(f1.x, f1.y);
                af.u[3] = pack_bf2(f1.z, f1.w);
                af.u[4] = pack_bf2(f2.x, f2.y);
                af.u[5] = pack_bf2(f2.z, f2.w);
                af.u[6] = pack_bf2(f3.x, f3.y);
                af.u[7] = pack_bf2(f3.z, f3.w);
            }
            FragBF bf;
            load_frag<68>(bf, kvt, ns3, lane, kk * 16);
            oacc = wmma_bf16(af, bf, oacc);
        }
        __syncthreads();  // kvt reused next chunk
    }

    const int d = ns3 + lo;
#pragma unroll
    for (int r = 0; r < 8; ++r) {
        const int row = mt3 * 16 + r + hi * 8;           // 0..31
        const int t = q0 + row;
        const float val = oacc[r] * rowinv[row];
        ohead[(size_t)(b * TT + t) * CC + h * DD + d] = f2bf(val);
    }
}

// ---------------------------------------------------------------------------
// Kernel 3: out = ohead(bf16) @ Wo + bo, fp32 [B,T,C].
// A tile (already bf16) arrives via async DMA; W converts on the VALU path.
// Grid: (8, 64), 256 threads.
// ---------------------------------------------------------------------------
__global__ __launch_bounds__(256) void out_proj(const unsigned short* __restrict__ Abf,
                                                const float* __restrict__ W,
                                                const float* __restrict__ bias,
                                                float* __restrict__ out) {
    __shared__ unsigned lds_a[128 * 20];
    __shared__ unsigned lds_b[128 * 20];
    const int tid = threadIdx.x;
    const int lane = tid & 31, wave = tid >> 5;
    const int wr = wave >> 2, wc = wave & 3;
    const int m0 = blockIdx.y * 128;
    const int n0 = blockIdx.x * 128;

    const unsigned* Ap = reinterpret_cast<const unsigned*>(Abf);  // bf16 pairs

    v8f acc[4][2];
#pragma unroll
    for (int mt = 0; mt < 4; ++mt)
#pragma unroll
        for (int nt = 0; nt < 2; ++nt) acc[mt][nt] = zero8();

    for (int k0 = 0; k0 < CC; k0 += 32) {
        if (k0 + 32 < CC) {
            __builtin_prefetch(W + (size_t)(k0 + 32 + 2 * (tid >> 7)) * CC + n0 + (tid & 127), 0, 3);
        }
        // A tile: 128 rows x 16 pairs, async DMA (2 b128 per thread)
#pragma unroll
        for (int i = 0; i < 2; ++i) {
            const int t = tid + i * 256;                 // 0..511
            const int row = t >> 2, c4 = (t & 3) * 4;
            async_ld_b128(Ap + (size_t)(m0 + row) * (CC / 2) + (k0 >> 1) + c4,
                          lds_a + row * 20 + c4);
        }
#pragma unroll
        for (int i = 0; i < 8; ++i) {
            const int idx = tid + i * 256;
            const int p = idx >> 7, n = idx & 127;
            const float w0 = W[(size_t)(k0 + 2 * p) * CC + n0 + n];
            const float w1 = W[(size_t)(k0 + 2 * p + 1) * CC + n0 + n];
            lds_b[n * 20 + p] = pack_bf2(w0, w1);
        }
        wait_async0();
        __syncthreads();

        FragBF bfr[2];
        load_frag<20>(bfr[0], lds_b, wc * 32 + 0, lane, 0);
        load_frag<20>(bfr[1], lds_b, wc * 32 + 16, lane, 0);
#pragma unroll
        for (int mt = 0; mt < 4; ++mt) {
            FragBF afr;
            load_frag<20>(afr, lds_a, wr * 64 + mt * 16, lane, 0);
            acc[mt][0] = wmma_bf16(afr, bfr[0], acc[mt][0]);
            acc[mt][1] = wmma_bf16(afr, bfr[1], acc[mt][1]);
        }
        __syncthreads();
    }

    const int lo = lane & 15, hi = lane >> 4;
#pragma unroll
    for (int nt = 0; nt < 2; ++nt) {
        const int n = n0 + wc * 32 + nt * 16 + lo;
        const float bv = bias[n];
#pragma unroll
        for (int mt = 0; mt < 4; ++mt) {
#pragma unroll
            for (int r = 0; r < 8; ++r) {
                const int m = m0 + wr * 64 + mt * 16 + r + hi * 8;
                out[(size_t)m * CC + n] = acc[mt][nt][r] + bv;
            }
        }
    }
}

// ---------------------------------------------------------------------------
extern "C" void kernel_launch(void* const* d_in, const int* in_sizes, int n_in,
                              void* d_out, int out_size, void* d_ws, size_t ws_size,
                              hipStream_t stream) {
    (void)in_sizes; (void)n_in; (void)out_size; (void)ws_size;

    const float* x  = (const float*)d_in[0];
    const float* Wq = (const float*)d_in[1];
    const float* bq = (const float*)d_in[2];
    const float* Wk = (const float*)d_in[3];
    const float* bk = (const float*)d_in[4];
    const float* Wv = (const float*)d_in[5];
    const float* bv = (const float*)d_in[6];
    const float* Wo = (const float*)d_in[7];
    const float* bo = (const float*)d_in[8];

    float* out  = (float*)d_out;                         // [B,T,C] fp32
    float* attn = out + (size_t)BB * TT * CC;            // [B,H,T,T] fp32

    unsigned short* Qw = (unsigned short*)d_ws;          // bf16 [B,H,T,D]
    unsigned short* Kw = Qw + (size_t)MM * CC;
    unsigned short* Vw = Kw + (size_t)MM * CC;
    unsigned short* Ow = Vw + (size_t)MM * CC;           // bf16 [B,T,C]

    dim3 blk(256);

    dim3 gProj(CC / 128, MM / 128);                      // (8, 64)
    qkv_proj<<<gProj, blk, 0, stream>>>(x, Wq, bq, Qw);
    qkv_proj<<<gProj, blk, 0, stream>>>(x, Wk, bk, Kw);
    qkv_proj<<<gProj, blk, 0, stream>>>(x, Wv, bv, Vw);

    // fused scores + softmax + attn-store + attn@V (279 KB dynamic LDS/WGP)
    const unsigned dynLds = (32 * SROW + 32 * 36 + 128 * 36) * sizeof(unsigned);
    (void)hipFuncSetAttribute(reinterpret_cast<const void*>(attn_fused),
                              hipFuncAttributeMaxDynamicSharedMemorySize,
                              (int)dynLds);
    dim3 gAtt(TT / 32, BB * HH);                         // (64, 64)
    attn_fused<<<gAtt, blk, dynLds, stream>>>(Qw, Kw, Vw, attn, Ow);

    out_proj<<<gProj, blk, 0, stream>>>(Ow, Wo, bo, out);
}
